// DynamicConv_46797963657413
// MI455X (gfx1250) — matrix operand order repaired and checked by reading, not compiled, not run on hardware
//
#include <hip/hip_runtime.h>
#include <hip/hip_bf16.h>

// MI455X (gfx1250) dynamic conv via SE kernel-generation + implicit-GEMM WMMA.
// Phases:
//   1) pool:  pooled[b][c] = mean_{hw} x[b][c][h][w]
//   2) fc1:   h = relu(pooled @ w1^T)              (16 x 32)
//   3) genw:  wh[b][oc][(ky*3+kx)*128+ic] = f16( h @ w2^T + b2 )   (f16 weights, GEMM K-layout)
//   4) xpose: xt[b][y*64+x][ic] = f16(x[b][ic][y][x])              (ic-minor for B fragments)
//   5) conv:  per (b, y-row) workgroup, 8 waves, v_wmma_f32_16x16x32_f16
// Workspace layout (needs ~20.5 MiB):
//   [0,      8KB)   pooled  (2048 f32)
//   [8KB,   16KB)   h       (512 f32)
//   [16KB, +16MB)   xt      (16*4096*128 f16)
//   [+16MB,+4.5MB)  wh      (16*147456 f16)

typedef __attribute__((ext_vector_type(16))) _Float16 v16h;
typedef __attribute__((ext_vector_type(8)))  _Float16 v8h;
typedef __attribute__((ext_vector_type(8)))  float    v8f;

#define B_N     16
#define C_IN    128
#define C_OUT   128
#define HW      4096
#define IMG     64
#define KDIM    1152      // 128 * 9
#define KWDIM   147456    // 128 * 1152
#define RED     32

// ---------------- phase 1: global average pool ----------------
__global__ __launch_bounds__(256)
void pool_kernel(const float* __restrict__ x, float* __restrict__ pooled) {
    __shared__ float red[256];
    const size_t base = (size_t)blockIdx.x * HW;   // blockIdx.x = b*128 + c
    float s = 0.0f;
    for (int i = threadIdx.x; i < HW; i += 256) s += x[base + i];
    red[threadIdx.x] = s;
    __syncthreads();
    for (int off = 128; off > 0; off >>= 1) {
        if (threadIdx.x < off) red[threadIdx.x] += red[threadIdx.x + off];
        __syncthreads();
    }
    if (threadIdx.x == 0) pooled[blockIdx.x] = red[0] * (1.0f / (float)HW);
}

// ---------------- phase 2: h = relu(pooled @ w1^T) ----------------
__global__ __launch_bounds__(512)
void fc1_kernel(const float* __restrict__ pooled, const float* __restrict__ w1,
                float* __restrict__ h) {
    __shared__ float p[B_N * C_IN];
    for (int i = threadIdx.x; i < B_N * C_IN; i += 512) p[i] = pooled[i];
    __syncthreads();
    const int b = threadIdx.x >> 5;   // 0..15
    const int r = threadIdx.x & 31;   // 0..31
    float acc = 0.0f;
    #pragma unroll 8
    for (int c = 0; c < C_IN; ++c) acc += p[b * C_IN + c] * w1[r * C_IN + c];
    h[b * RED + r] = fmaxf(acc, 0.0f);
}

// ---------------- phase 3: generate per-sample f16 conv weights ----------------
// storage index s = oc*1152 + tap*128 + ic  maps to  k_ref = oc*1152 + ic*9 + tap
__global__ __launch_bounds__(256)
void genw_kernel(const float* __restrict__ h, const float* __restrict__ w2,
                 const float* __restrict__ b2, _Float16* __restrict__ wh) {
    __shared__ float hs[B_N * RED];
    hs[threadIdx.x]       = h[threadIdx.x];
    hs[threadIdx.x + 256] = h[threadIdx.x + 256];
    __syncthreads();

    const int s   = blockIdx.x * 256 + threadIdx.x;   // 0..147455
    const int oc  = s / KDIM;
    const int rem = s - oc * KDIM;
    const int tap = rem >> 7;     // 0..8
    const int ic  = rem & 127;
    const int k_ref = oc * KDIM + ic * 9 + tap;

    const float4* wrow = (const float4*)(w2 + (size_t)k_ref * RED);  // 128B aligned
    float4 w[8];
    #pragma unroll
    for (int i = 0; i < 8; ++i) w[i] = wrow[i];
    const float bias = b2[k_ref];

    #pragma unroll
    for (int b = 0; b < B_N; ++b) {
        const float* hb = &hs[b * RED];
        float acc = bias;
        #pragma unroll
        for (int i = 0; i < 8; ++i)
            acc += w[i].x * hb[4*i] + w[i].y * hb[4*i+1] +
                   w[i].z * hb[4*i+2] + w[i].w * hb[4*i+3];
        wh[(size_t)b * KWDIM + s] = (_Float16)acc;
    }
}

// ---------------- phase 4: x -> xt[b][pixel][ic] f16 (LDS tile transpose) ----------------
#define TSTR 132   // half-stride per col: 264B = 66 dwords -> +2 banks per col, conflict-free
__global__ __launch_bounds__(256)
void xpose_kernel(const float* __restrict__ x, _Float16* __restrict__ xt) {
    __shared__ __align__(16) _Float16 tile[IMG * TSTR];
    const int b = blockIdx.x >> 6;
    const int y = blockIdx.x & 63;
    const int t = threadIdx.x;

    const int col = t & 63;
    const int ics = t >> 6;       // 0..3
    for (int icb = 0; icb < C_IN; icb += 4) {
        const int ic = icb + ics;
        const float v = x[(((size_t)b * C_IN + ic) * IMG + y) * IMG + col];
        tile[col * TSTR + ic] = (_Float16)v;
    }
    __syncthreads();

    const int ic2 = t & 127;
    const int cs  = t >> 7;       // 0..1
    for (int c0 = 0; c0 < IMG; c0 += 2) {
        const int c = c0 + cs;
        xt[((size_t)b * HW + y * IMG + c) * C_IN + ic2] = tile[c * TSTR + ic2];
    }
}

// ---------------- phase 5: per-sample implicit-GEMM conv with WMMA ----------------
// GEMM per sample: D(128 oc x 4096 px) = W(128 x 1152) * im2col(1152 x 4096)
// One WG per (b, image row y). 8 waves; wave w owns oc block w*16..w*16+15, all 64 px.
#define ICS 136    // half-stride per (row,col) LDS cell: 272B = 68 dwords -> +4 banks/col
__global__ __launch_bounds__(256)
void conv_kernel(const _Float16* __restrict__ xt, const _Float16* __restrict__ wh,
                 float* __restrict__ out) {
    __shared__ __align__(16) _Float16 xs[3 * 66 * ICS];   // 53,856 B

    const int b = blockIdx.x >> 6;
    const int y = blockIdx.x & 63;
    const int t = threadIdx.x;

    // ---- stage 3-row halo tile: xs[row][col][ic], col 0 == image col -1 ----
    if (t < 3 * 66) {
        const int row = t / 66, col = t % 66;
        const int sy = y + row - 1, sx = col - 1;
        uint4* d4 = (uint4*)&xs[(row * 66 + col) * ICS];   // cell offset = 272B, 16B aligned
        if (sy >= 0 && sy < IMG && sx >= 0 && sx < IMG) {
            const uint4* s4 = (const uint4*)&xt[((size_t)b * HW + sy * IMG + sx) * C_IN];
            #pragma unroll
            for (int i = 0; i < 16; ++i) d4[i] = s4[i];
        } else {
            const uint4 z = {0u, 0u, 0u, 0u};
            #pragma unroll
            for (int i = 0; i < 16; ++i) d4[i] = z;
        }
    }
    __syncthreads();

    const int wave = t >> 5;
    const int lane = t & 31;
    const int ln   = lane & 15;
    const int hi   = lane >> 4;
    const int oc_a = wave * 16 + ln;            // A-matrix row for this lane

    const _Float16* wbase = wh + ((size_t)b * KWDIM + (size_t)oc_a * KDIM);

    v8f acc[4] = {};   // 4 N-blocks of 16 pixels

    for (int tap = 0; tap < 9; ++tap) {
        const int ky = tap / 3, kx = tap - 3 * (tap / 3);
        for (int icb = 0; icb < 4; ++icb) {
            const int k0 = tap * 128 + icb * 32;

            // A fragment per ISA 16-bit A 16x32 table:
            //   halves 0..7  -> k = 8*hi + 0..7
            //   halves 8..15 -> k = 16 + 8*hi + 0..7
            v8h alo = *(const v8h*)(wbase + k0 + 8 * hi);
            v8h ahi = *(const v8h*)(wbase + k0 + 16 + 8 * hi);
            v16h a = __builtin_shufflevector(alo, ahi,
                        0,1,2,3,4,5,6,7,8,9,10,11,12,13,14,15);

            if (k0 < 1120)   // uniform branch: prefetch next K-block of weights
                __builtin_prefetch(wbase + k0 + 32, 0, 1);

            #pragma unroll
            for (int nb = 0; nb < 4; ++nb) {
                // B fragment (K x N = 32 x 16): lane = col (ln), same K table as A:
                //   halves 0..7 -> k = 8*hi + 0..7, halves 8..15 -> k = 16 + 8*hi + 0..7
                const _Float16* bsrc =
                    &xs[(ky * 66 + nb * 16 + ln + kx) * ICS + icb * 32 + 8 * hi];
                v8h blo = *(const v8h*)bsrc;
                v8h bhi = *(const v8h*)(bsrc + 16);
                v16h bm = __builtin_shufflevector(blo, bhi,
                            0,1,2,3,4,5,6,7,8,9,10,11,12,13,14,15);

                acc[nb] = __builtin_amdgcn_wmma_f32_16x16x32_f16(
                              false, a, false, bm, (short)0, acc[nb], false, false);
            }
        }
    }

    // ---- store D: lane holds N=ln, M = r + 8*hi per VGPR r ----
    #pragma unroll
    for (int nb = 0; nb < 4; ++nb) {
        #pragma unroll
        for (int r = 0; r < 8; ++r) {
            const int ocr = wave * 16 + r + 8 * hi;
            const int px  = nb * 16 + ln;
            out[(((size_t)b * C_OUT + ocr) * IMG + y) * IMG + px] = acc[nb][r];
        }
    }
}

// ---------------- launcher ----------------
extern "C" void kernel_launch(void* const* d_in, const int* in_sizes, int n_in,
                              void* d_out, int out_size, void* d_ws, size_t ws_size,
                              hipStream_t stream) {
    const float* x  = (const float*)d_in[0];
    const float* w1 = (const float*)d_in[1];
    const float* w2 = (const float*)d_in[2];
    const float* b2 = (const float*)d_in[3];
    float* out = (float*)d_out;

    char* ws = (char*)d_ws;
    float*    pooled = (float*)(ws);                                  // 8 KB
    float*    h      = (float*)(ws + 8192);                           // 2 KB
    _Float16* xt     = (_Float16*)(ws + 16384);                       // 16 MB
    _Float16* wh     = (_Float16*)(ws + 16384 + (size_t)B_N*HW*C_IN*2); // 4.5 MB

    pool_kernel<<<B_N * C_IN, 256, 0, stream>>>(x, pooled);
    fc1_kernel<<<1, 512, 0, stream>>>(pooled, w1, h);
    genw_kernel<<<KWDIM / 256, 256, 0, stream>>>(h, w2, b2, wh);
    xpose_kernel<<<B_N * IMG, 256, 0, stream>>>(x, xt);
    conv_kernel<<<B_N * IMG, 256, 0, stream>>>(xt, wh, out);
}
